// VectorQuantizer_31267361915564
// MI455X (gfx1250) — matrix operand order repaired and checked by reading, not compiled, not run on hardware
//
#include <hip/hip_runtime.h>

typedef __attribute__((ext_vector_type(16))) __bf16       v16bf;
typedef __attribute__((ext_vector_type(8)))  float        v8f;
typedef __attribute__((ext_vector_type(4)))  unsigned int v4u;

#define B_   64
#define C_   256
#define HW_  1024          // 32*32
#define N_   65536         // B_*HW_
#define K_   1024          // codebook entries
#define CHW  262144        // C_*HW_

static __device__ __forceinline__ unsigned short f32_to_bf16(float f) {
    unsigned int u = __float_as_uint(f);
    unsigned int r = u + 0x7FFFu + ((u >> 16) & 1u);   // round-to-nearest-even
    return (unsigned short)(r >> 16);
}

// ---------------------------------------------------------------------------
// Kernel 1: z (B,C,H,W) fp32  ->  zb (N=B*HW, C) row-major bf16 (LDS transpose)
// ---------------------------------------------------------------------------
__global__ void __launch_bounds__(256)
k_transpose_z_bf16(const float* __restrict__ z, unsigned short* __restrict__ zb) {
    __shared__ float tile[32][33];
    int blk = blockIdx.x;            // 64 * 32 * 8 = 16384 blocks
    int c0  = (blk & 7) * 32;
    int hw0 = ((blk >> 3) & 31) * 32;
    int b   = blk >> 8;
    int tid = threadIdx.x;

    const float* src = z + (size_t)b * CHW;
#pragma unroll
    for (int it = 0; it < 4; ++it) {
        int i = it * 256 + tid;
        int cl = i >> 5, hwl = i & 31;
        tile[cl][hwl] = src[(c0 + cl) * HW_ + hw0 + hwl];   // coalesced along hw
    }
    __syncthreads();
    unsigned short* dst = zb + (size_t)b * HW_ * C_;
#pragma unroll
    for (int it = 0; it < 4; ++it) {
        int i = it * 256 + tid;
        int hwl = i >> 5, cl = i & 31;
        dst[(size_t)(hw0 + hwl) * C_ + c0 + cl] = f32_to_bf16(tile[cl][hwl]); // coalesced along c
    }
}

// ---------------------------------------------------------------------------
// Kernel 2: emb fp32 -> bf16 + per-row squared norms
// ---------------------------------------------------------------------------
__global__ void __launch_bounds__(256)
k_emb_prep(const float* __restrict__ emb, unsigned short* __restrict__ ebf,
           float* __restrict__ norms) {
    __shared__ float red[256];
    int n = blockIdx.x, tid = threadIdx.x;
    float e = emb[n * C_ + tid];
    ebf[n * C_ + tid] = f32_to_bf16(e);
    red[tid] = e * e;
    __syncthreads();
    for (int s = 128; s > 0; s >>= 1) {
        if (tid < s) red[tid] += red[tid + s];
        __syncthreads();
    }
    if (tid == 0) norms[n] = red[0];
}

// ---------------------------------------------------------------------------
// Kernel 3: WMMA distance GEMM + argmin.
// One wave handles TWO 16-row M tiles (32 rows): A fragments for both tiles
// register-resident; each B fragment feeds two independent WMMA accumulator
// chains, and B fragments are explicitly double-buffered so the next load
// overlaps the current WMMA pair (and the argmin tail overlaps the first
// load of the next codebook tile).
// d = ||e||^2 - 2 z.e   (||z||^2 dropped: constant per row)
// ---------------------------------------------------------------------------
__global__ void __launch_bounds__(256)
k_vq_argmin(const unsigned short* __restrict__ zb,
            const unsigned short* __restrict__ ebf,
            const float* __restrict__ norms,
            int* __restrict__ idx_out) {
    int lane    = threadIdx.x & 31;
    int wave    = threadIdx.x >> 5;
    int m0      = (blockIdx.x * 8 + wave) * 32;   // rows m0..m0+31
    int lm      = lane & 15;
    int halfSel = lane >> 4;

    // A fragments: 16-bit 16x32 layout -> per lane two contiguous 16B chunks
    union Frag { v16bf v; v4u u[2]; };
    Frag a0[8], a1[8];
    const unsigned short* arow0 = zb + (size_t)(m0 + lm) * C_;
    const unsigned short* arow1 = arow0 + 16 * C_;
#pragma unroll
    for (int kk = 0; kk < 8; ++kk) {
        int k0 = kk * 32 + halfSel * 8;
        a0[kk].u[0] = *(const v4u*)(arow0 + k0);
        a0[kk].u[1] = *(const v4u*)(arow0 + k0 + 16);
        a1[kk].u[0] = *(const v4u*)(arow1 + k0);
        a1[kk].u[1] = *(const v4u*)(arow1 + k0 + 16);
    }

    float bestv0[8], bestv1[8];
    int   bestn0[8], bestn1[8];
#pragma unroll
    for (int v = 0; v < 8; ++v) {
        bestv0[v] = 3.4e38f; bestn0[v] = 0;
        bestv1[v] = 3.4e38f; bestn1[v] = 0;
    }

    // B fragment base for this lane: 32x16 layout -> lane holds two 16B chunks
    // of emb row (n0 + lm) at k = 8*half and 16 + 8*half (same as A).
    const unsigned short* browBase = ebf + (size_t)lm * C_ + halfSel * 8;

    // Prime the pipeline: first fragment of tile 0 into buffer 0.
    Frag bf[2];
    bf[0].u[0] = *(const v4u*)(browBase);
    bf[0].u[1] = *(const v4u*)(browBase + 16);

    for (int nt = 0; nt < 64; ++nt) {
        const unsigned short* brow     = browBase + (size_t)(nt * 16) * C_;
        const unsigned short* browNext = brow + (size_t)16 * C_;
        v8f c0 = {};
        v8f c1 = {};
#pragma unroll
        for (int kk = 0; kk < 8; ++kk) {
            int cur = kk & 1;
            int nxt = cur ^ 1;
            if (kk < 7) {                       // prefetch next fragment of this tile
                bf[nxt].u[0] = *(const v4u*)(brow + (kk + 1) * 32);
                bf[nxt].u[1] = *(const v4u*)(brow + (kk + 1) * 32 + 16);
            } else if (nt < 63) {               // prefetch first fragment of next tile
                bf[nxt].u[0] = *(const v4u*)(browNext);
                bf[nxt].u[1] = *(const v4u*)(browNext + 16);
            }
            c0 = __builtin_amdgcn_wmma_f32_16x16x32_bf16(false, a0[kk].v, false, bf[cur].v,
                                                         (short)0, c0, false, false);
            c1 = __builtin_amdgcn_wmma_f32_16x16x32_bf16(false, a1[kk].v, false, bf[cur].v,
                                                         (short)0, c1, false, false);
        }
        int   ncand = nt * 16 + lm;
        float ne    = norms[ncand];
#pragma unroll
        for (int v = 0; v < 8; ++v) {
            float d0 = ne - 2.0f * c0[v];
            if (d0 < bestv0[v]) { bestv0[v] = d0; bestn0[v] = ncand; }
            float d1 = ne - 2.0f * c1[v];
            if (d1 < bestv1[v]) { bestv1[v] = d1; bestn1[v] = ncand; }
        }
    }

    // reduce across the 16 lanes sharing each m (halves hold m+v and m+8+v)
#pragma unroll
    for (int v = 0; v < 8; ++v) {
#pragma unroll
        for (int off = 1; off < 16; off <<= 1) {
            float ov0 = __shfl_xor(bestv0[v], off, 32);
            int   on0 = __shfl_xor(bestn0[v], off, 32);
            if (ov0 < bestv0[v] || (ov0 == bestv0[v] && on0 < bestn0[v])) {
                bestv0[v] = ov0; bestn0[v] = on0;
            }
            float ov1 = __shfl_xor(bestv1[v], off, 32);
            int   on1 = __shfl_xor(bestn1[v], off, 32);
            if (ov1 < bestv1[v] || (ov1 == bestv1[v] && on1 < bestn1[v])) {
                bestv1[v] = ov1; bestn1[v] = on1;
            }
        }
    }
    if (lm == 0) {
        int mbase = m0 + halfSel * 8;
#pragma unroll
        for (int v = 0; v < 8; ++v) {
            idx_out[mbase + v]      = bestn0[v];
            idx_out[mbase + 16 + v] = bestn1[v];
        }
    }
}

// ---------------------------------------------------------------------------
// Kernel 4: gather emb[idx] -> out (B,C,H,W), loss = 1.25*mean((e - z)^2),
// indices written as float.  Output-major indexing: z/out accesses coalesced,
// emb gather hits L2/WGP$ (1 MB).
// ---------------------------------------------------------------------------
__global__ void __launch_bounds__(256)
k_finalize(const float* __restrict__ z, const float* __restrict__ emb,
           const int* __restrict__ idx_ws, float* __restrict__ out,
           float* __restrict__ loss, float* __restrict__ idx_f) {
    __shared__ float red[256];
    int tid = threadIdx.x;
    size_t gid = (size_t)blockIdx.x * 256 + tid;
    int b  = (int)(gid >> 18);
    int c  = (int)((gid >> 10) & 255);
    int hw = (int)(gid & 1023);
    int n  = b * HW_ + hw;

    int   id = idx_ws[n];
    float e  = emb[id * C_ + c];
    float zv = z[gid];
    out[gid] = e;
    if (c == 0) idx_f[n] = (float)id;

    float d = e - zv;
    red[tid] = d * d;
    __syncthreads();
    for (int s = 128; s > 0; s >>= 1) {
        if (tid < s) red[tid] += red[tid + s];
        __syncthreads();
    }
    if (tid == 0) atomicAdd(loss, red[0] * (1.25f / 16777216.0f));
}

// ---------------------------------------------------------------------------
extern "C" void kernel_launch(void* const* d_in, const int* in_sizes, int n_in,
                              void* d_out, int out_size, void* d_ws, size_t ws_size,
                              hipStream_t stream) {
    const float* z   = (const float*)d_in[0];   // (64,256,32,32) fp32
    const float* emb = (const float*)d_in[1];   // (1024,256)     fp32

    unsigned char* ws = (unsigned char*)d_ws;
    unsigned short* zb  = (unsigned short*)ws;                 // 33,554,432 B  (N x C bf16)
    unsigned short* ebf = (unsigned short*)(ws + 33554432u);   //    524,288 B  (K x C bf16)
    float* norms        = (float*)(ws + 34078720u);            //      4,096 B
    int*   idxw         = (int*)(ws + 34082816u);              //    262,144 B

    float* out  = (float*)d_out;        // 16,777,216 elems
    float* loss = out + 16777216;       // 1 elem
    float* idxf = loss + 1;             // 65,536 elems (indices as float)

    k_transpose_z_bf16<<<16384, 256, 0, stream>>>(z, zb);
    k_emb_prep<<<1024, 256, 0, stream>>>(emb, ebf, norms);
    hipMemsetAsync(loss, 0, sizeof(float), stream);
    k_vq_argmin<<<256, 256, 0, stream>>>(zb, ebf, norms, idxw);
    k_finalize<<<65536, 256, 0, stream>>>(z, emb, idxw, out, loss, idxf);
}
